// NodeCentric_15479062134971
// MI455X (gfx1250) — matrix operand order repaired and checked by reading, not compiled, not run on hardware
//
#include <hip/hip_runtime.h>

typedef __attribute__((ext_vector_type(16))) _Float16     v16h;
typedef __attribute__((ext_vector_type(4)))  _Float16     v4h;
typedef __attribute__((ext_vector_type(8)))  float        v8f;
typedef __attribute__((ext_vector_type(8)))  unsigned int v8u;

#define FXd 128   // node in-features
#define FEd 32    // edge in-features
#define OXd 128   // node out-features
#define OEd 128   // edge out-features

// ---------------- zero agg ----------------
__global__ __launch_bounds__(256) void k_zero(float* __restrict__ p, int n4) {
  int i = blockIdx.x * 256 + threadIdx.x;
  if (i < n4) ((float4*)p)[i] = make_float4(0.f, 0.f, 0.f, 0.f);
}

// ------------- segment-sum scatter (E x FE) -------------
__global__ __launch_bounds__(256) void k_scatter(const float* __restrict__ ea,
                                                 const int*   __restrict__ rows,
                                                 float* __restrict__ agg, int E) {
  int t = blockIdx.x * 256 + threadIdx.x;
  int e = t >> 3;
  int c = (t & 7) * 4;
  if (e >= E) return;
  int r = rows[e];                       // edge_index[0][e]
  const float4 v = *(const float4*)(ea + (size_t)e * FEd + c);
  float* dst = agg + (size_t)r * FEd + c;
  unsafeAtomicAdd(dst + 0, v.x);
  unsafeAtomicAdd(dst + 1, v.y);
  unsafeAtomicAdd(dst + 2, v.z);
  unsafeAtomicAdd(dst + 3, v.w);
}

// ------------- weight transpose + f32->f16 convert -------------
// WxT[n][k] = Wx[k][n]  (OXd x FXd),  WeT[n][k] = We[k][n]  (OEd x FEd)
__global__ __launch_bounds__(256) void k_cvt(const float* __restrict__ Wx,
                                             const float* __restrict__ We,
                                             _Float16* __restrict__ WxT,
                                             _Float16* __restrict__ WeT) {
  int t = blockIdx.x * 256 + threadIdx.x;
  if (t < FXd * OXd) {
    int n = t / FXd, k = t % FXd;
    WxT[t] = (_Float16)Wx[k * OXd + n];
  } else {
    int u = t - FXd * OXd;
    if (u < FEd * OEd) {
      int n = u / FEd, k = u % FEd;
      WeT[u] = (_Float16)We[k * OEd + n];
    }
  }
}

// ------------- fused dual-GEMM + bias + relu + concat -------------
// grid.x = ceil(N/16); 256 threads = 8 waves; wave w owns output cols [16w,16w+16)
// of BOTH hx (out cols 0..127) and he (out cols 128..255).
__global__ __launch_bounds__(256) void k_fused(const float* __restrict__ x,
                                               const float* __restrict__ agg,
                                               const _Float16* __restrict__ WxT,
                                               const _Float16* __restrict__ WeT,
                                               const float* __restrict__ bx,
                                               const float* __restrict__ be,
                                               float* __restrict__ out, int N) {
  __shared__ __align__(16) _Float16 xs[16 * FXd];           // x tile, f16 (4 KB)
  __shared__ __align__(16) _Float16 as[16 * FEd];           // agg tile, f16 (1 KB)
  __shared__ __align__(16) float    outs[16 * (OXd + OEd)]; // 16x256 f32 (16 KB)

  const int tid  = threadIdx.x;
  const int row0 = blockIdx.x * 16;

  // stage tiles into LDS: float4 loads, packed f16x4 (8B) LDS stores
#pragma unroll
  for (int i = tid; i < 16 * FXd / 4; i += 256) {   // 512 quads -> 2 per thread
    int e4 = i * 4;
    int m  = e4 >> 7;
    int r  = row0 + m;
    float4 v = (r < N) ? *(const float4*)(x + (size_t)r * FXd + (e4 & (FXd - 1)))
                       : make_float4(0.f, 0.f, 0.f, 0.f);
    v4h hv = { (_Float16)v.x, (_Float16)v.y, (_Float16)v.z, (_Float16)v.w };
    *(v4h*)(xs + e4) = hv;
  }
  {                                                  // 128 quads -> threads 0..127
    int i = tid;
    if (i < 16 * FEd / 4) {
      int e4 = i * 4;
      int m  = e4 >> 5;
      int r  = row0 + m;
      float4 v = (r < N) ? *(const float4*)(agg + (size_t)r * FEd + (e4 & (FEd - 1)))
                         : make_float4(0.f, 0.f, 0.f, 0.f);
      v4h hv = { (_Float16)v.x, (_Float16)v.y, (_Float16)v.z, (_Float16)v.w };
      *(v4h*)(as + e4) = hv;
    }
  }
  __syncthreads();

  const int wave = tid >> 5;     // 0..7  -> 16-column slab
  const int lane = tid & 31;
  const int mn   = lane & 15;    // row (A) / col (B,D)
  const int h    = lane >> 4;    // K-half selector

  // ---------- hx = x @ Wx : 4 WMMA steps over K=128, 2 independent chains ----------
  v8f acc0 = {}, acc1 = {};
  const _Float16* wcol = WxT + (size_t)(wave * 16 + mn) * FXd; // contiguous column of Wx
  const _Float16* arow = xs + mn * FXd;
#pragma unroll
  for (int ks = 0; ks < FXd / 32; ++ks) {
    v8u au, bu;
#pragma unroll
    for (int i = 0; i < 8; ++i) {
      // ISA A-layout (16-bit 16x32): v0..3 -> K {0..7}+8h, v4..7 -> K {16..23}+8h
      int ka = (i < 4) ? (2 * i + 8 * h) : (16 + 2 * (i - 4) + 8 * h);
      // B-layout matched to A: v0..3 -> K {0..7}+8h, v4..7 -> K {16..23}+8h
      int kb = (i < 4) ? (2 * i + 8 * h) : (16 + 2 * (i - 4) + 8 * h);
      au[i] = *(const unsigned int*)(arow + ks * 32 + ka);  // ka even -> 4B aligned
      bu[i] = *(const unsigned int*)(wcol + ks * 32 + kb);
    }
    if (ks & 1)
      acc1 = __builtin_amdgcn_wmma_f32_16x16x32_f16(
          false, __builtin_bit_cast(v16h, au),
          false, __builtin_bit_cast(v16h, bu),
          (short)0, acc1, false, false);
    else
      acc0 = __builtin_amdgcn_wmma_f32_16x16x32_f16(
          false, __builtin_bit_cast(v16h, au),
          false, __builtin_bit_cast(v16h, bu),
          (short)0, acc0, false, false);
  }

  // ---------- he = agg @ We : single WMMA, K=32 ----------
  v8f acc2 = {};
  {
    const _Float16* wecol = WeT + (size_t)(wave * 16 + mn) * FEd;
    const _Float16* arow2 = as + mn * FEd;
    v8u au, bu;
#pragma unroll
    for (int i = 0; i < 8; ++i) {
      int ka = (i < 4) ? (2 * i + 8 * h) : (16 + 2 * (i - 4) + 8 * h);
      int kb = (i < 4) ? (2 * i + 8 * h) : (16 + 2 * (i - 4) + 8 * h);
      au[i] = *(const unsigned int*)(arow2 + ka);
      bu[i] = *(const unsigned int*)(wecol + kb);
    }
    acc2 = __builtin_amdgcn_wmma_f32_16x16x32_f16(
        false, __builtin_bit_cast(v16h, au),
        false, __builtin_bit_cast(v16h, bu),
        (short)0, acc2, false, false);
  }

  // bias + relu into the LDS output tile (D layout: m = vgpr + 8*half, n = lane%16)
  const float bh  = bx[wave * 16 + mn];
  const float bh2 = be[wave * 16 + mn];
#pragma unroll
  for (int r = 0; r < 8; ++r) {
    int m = r + 8 * h;
    outs[m * (OXd + OEd) + wave * 16 + mn]       = fmaxf(acc0[r] + acc1[r] + bh, 0.f);
    outs[m * (OXd + OEd) + OXd + wave * 16 + mn] = fmaxf(acc2[r] + bh2, 0.f);
  }
  __syncthreads();

  // coalesced writeback: thread t owns row t>>4, 16 consecutive floats at col (t&15)*16
  {
    const int m  = tid >> 4;
    const int c0 = (tid & 15) * 16;
    if (row0 + m < N) {
      float* dst = out + (size_t)(row0 + m) * (OXd + OEd) + c0;
      const float* src = outs + m * (OXd + OEd) + c0;
#pragma unroll
      for (int j = 0; j < 4; ++j)
        *(float4*)(dst + 4 * j) = *(const float4*)(src + 4 * j);
    }
  }
}

extern "C" void kernel_launch(void* const* d_in, const int* in_sizes, int n_in,
                              void* d_out, int out_size, void* d_ws, size_t ws_size,
                              hipStream_t stream) {
  const float* x  = (const float*)d_in[0];
  const int*   ei = (const int*)d_in[1];   // [2, E]; row index = first E entries
  const float* ea = (const float*)d_in[2];
  const float* Wx = (const float*)d_in[3];
  const float* bx = (const float*)d_in[4];
  const float* We = (const float*)d_in[5];
  const float* be = (const float*)d_in[6];
  float* out = (float*)d_out;

  const int N = in_sizes[0] / FXd;   // 100000
  const int E = in_sizes[2] / FEd;   // 1600000

  // workspace layout: agg f32 [N*FE] | WxT f16 [OX*FX] | WeT f16 [OE*FE]
  float*    agg = (float*)d_ws;
  _Float16* WxT = (_Float16*)((char*)d_ws + (size_t)N * FEd * sizeof(float));
  _Float16* WeT = WxT + (size_t)OXd * FXd;

  const int n4 = N * FEd / 4;
  k_zero<<<(n4 + 255) / 256, 256, 0, stream>>>(agg, n4);
  k_scatter<<<(E * 8 + 255) / 256, 256, 0, stream>>>(ea, ei, agg, E);
  k_cvt<<<(FXd * OXd + FEd * OEd + 255) / 256, 256, 0, stream>>>(Wx, We, WxT, WeT);
  k_fused<<<(N + 15) / 16, 256, 0, stream>>>(x, agg, WxT, WeT, bx, be, out, N);
}
